// ORCDF_Extractor_69879117906660
// MI455X (gfx1250) — matrix operand order repaired and checked by reading, not compiled, not run on hardware
//
#include <hip/hip_runtime.h>

#define STU 12000
#define EXERN 8000
#define KNOWN 600
#define NTOT (STU + EXERN + KNOWN)
#define DM 64
#define ND ((size_t)NTOT * DM)
#define LAMW 0.5f
#define SSLW 0.1f
#define INV_TEMP 1.25f   // 1/0.8

typedef __attribute__((ext_vector_type(16))) __bf16 bf16x16;
typedef __attribute__((ext_vector_type(8)))  float  f32x8;

union FragB { bf16x16 v; unsigned int u[8]; };

__device__ __forceinline__ unsigned short f2bf(float f) {
    union { float f; unsigned int u; } v; v.f = f;
    unsigned int r = v.u + 0x7FFFu + ((v.u >> 16) & 1u);  // RNE
    return (unsigned short)(r >> 16);
}
__device__ __forceinline__ unsigned int pack2(float a, float b) {
    return (unsigned int)f2bf(a) | ((unsigned int)f2bf(b) << 16);
}

// ---------------------------------------------------------------------------
// Generic WMMA GEMM: C[M x Nout] = act(A[M x K] @ W[K x Nout] + bias)
// A may be split across two buffers (concat path): k < Ksplit -> A0, else A1.
// Block: 256 thr (8 waves). Tile 64x64, K-stage 64 (two wmma k-steps),
// wave computes a 16x32 sub-tile -> 4 wmma per stage per wave.
// All K used here (768/512/256/128) are multiples of 64.
// ---------------------------------------------------------------------------
__global__ __launch_bounds__(256) void gemm_bias_act(
    const float* __restrict__ A0, const float* __restrict__ A1,
    int Ksplit, int lda0, int lda1,
    const float* __restrict__ Wt, const float* __restrict__ bias,
    const float* __restrict__ slope,
    float* __restrict__ C, int M, int K, int Nout)
{
    __shared__ unsigned short As[64][64];   // A tile, row-major bf16 (8 KB)
    __shared__ unsigned short Bs[64][64];   // B tile transposed: Bs[n][k] (8 KB)
    const int tid  = threadIdx.x;
    const int lane = tid & 31, wid = tid >> 5;
    const int lh   = lane & 15, hh = lane >> 4;
    const int mbase = blockIdx.y * 64;
    const int nbase = blockIdx.x * 64;
    const int wm = (wid & 3) * 16;   // 4 waves along M
    const int wn = (wid >> 2) * 32;  // 2 waves along N, 2 tiles each

    f32x8 c0 = {}; f32x8 c1 = {};
    const bool act  = (slope != nullptr);
    const float alp = act ? slope[0] : 0.f;

    for (int k0 = 0; k0 < K; k0 += 64) {
        if (k0 + 64 < K) __builtin_prefetch(&Wt[(size_t)(k0 + 64) * Nout + nbase], 0, 0);
        // cooperative A tile load (coalesced over k), fp32 -> bf16
        for (int e = tid; e < 64 * 64; e += 256) {
            int m = e >> 6, k = e & 63;
            int gm = mbase + m, gk = k0 + k;
            float v = 0.f;
            if (gm < M)
                v = (A1 == nullptr || gk < Ksplit)
                      ? A0[(size_t)gm * lda0 + gk]
                      : A1[(size_t)gm * lda1 + (gk - Ksplit)];
            As[m][k] = f2bf(v);
        }
        // cooperative B tile load (coalesced over n), stored transposed
        for (int e = tid; e < 64 * 64; e += 256) {
            int k = e >> 6, n = e & 63;
            Bs[n][k] = f2bf(Wt[(size_t)(k0 + k) * Nout + nbase + n]);
        }
        __syncthreads();

        FragB fa0, fa1, fb00, fb01, fb10, fb11;
        {   // A 16x32 bf16 layout: half hh -> K 16*hh..16*hh+15 of each k-step
            const unsigned int* ar = (const unsigned int*)&As[wm + lh][0];
            const int kb2 = hh * 8;
            #pragma unroll
            for (int i = 0; i < 4; i++) {
                fa0.u[i] = ar[kb2 + i];      fa0.u[4 + i] = ar[4 + kb2 + i];
                fa1.u[i] = ar[16 + kb2 + i]; fa1.u[4 + i] = ar[20 + kb2 + i];
            }
        }
        {   // B 32x16: lane half -> K 0-15 / 16-31 of each k-step
            const unsigned int* b0 = (const unsigned int*)&Bs[wn + lh][0];
            const unsigned int* b1 = (const unsigned int*)&Bs[wn + 16 + lh][0];
            const int kb2 = hh * 8;
            #pragma unroll
            for (int j = 0; j < 8; j++) {
                fb00.u[j] = b0[kb2 + j]; fb01.u[j] = b0[16 + kb2 + j];
                fb10.u[j] = b1[kb2 + j]; fb11.u[j] = b1[16 + kb2 + j];
            }
        }
        c0 = __builtin_amdgcn_wmma_f32_16x16x32_bf16(false, fa0.v, false, fb00.v, (short)0, c0, false, false);
        c1 = __builtin_amdgcn_wmma_f32_16x16x32_bf16(false, fa0.v, false, fb10.v, (short)0, c1, false, false);
        c0 = __builtin_amdgcn_wmma_f32_16x16x32_bf16(false, fa1.v, false, fb01.v, (short)0, c0, false, false);
        c1 = __builtin_amdgcn_wmma_f32_16x16x32_bf16(false, fa1.v, false, fb11.v, (short)0, c1, false, false);
        __syncthreads();
    }
    // C layout: lanes 0-15 N=lh, VGPR r -> M=r; lanes 16-31 -> M=8+r
    const int n0 = nbase + wn + lh, n1 = n0 + 16;
    const float b0v = bias[n0], b1v = bias[n1];
    #pragma unroll
    for (int r = 0; r < 8; r++) {
        int mrow = mbase + wm + hh * 8 + r;
        if (mrow < M) {
            float v0 = c0[r] + b0v;
            if (act) v0 = v0 >= 0.f ? v0 : alp * v0;
            C[(size_t)mrow * Nout + n0] = v0;
            float v1 = c1[r] + b1v;
            if (act) v1 = v1 >= 0.f ? v1 : alp * v1;
            C[(size_t)mrow * Nout + n1] = v1;
        }
    }
}

// ---------------------------------------------------------------------------
// SpMM: out[row] += val * emb[col], 64 threads per edge.
// Edge id is wave-uniform -> readfirstlane so row/col/val become s_loads.
// Relaxed agent-scope fetch_add -> native global_atomic_add_f32.
// ---------------------------------------------------------------------------
__global__ __launch_bounds__(256) void spmm_kernel(
    const int* __restrict__ idx, const float* __restrict__ val,
    const float* __restrict__ emb, float* __restrict__ outp, int E_)
{
    int gid = blockIdx.x * 256 + threadIdx.x;
    int e = __builtin_amdgcn_readfirstlane(gid >> 6);   // uniform per wave
    if (e >= E_) return;
    int d = gid & 63;
    int row = idx[e];          // idx[0][e]   (scalar load)
    int col = idx[E_ + e];     // idx[1][e]   (scalar load)
    float v = val[e];          //             (scalar load)
    float contrib = v * emb[(size_t)col * DM + d];
    __hip_atomic_fetch_add(&outp[(size_t)row * DM + d], contrib,
                           __ATOMIC_RELAXED, __HIP_MEMORY_SCOPE_AGENT);
}

// ---------------------------------------------------------------------------
// InfoNCE partial: accum += sum_i (logsumexp_j(v1_i . v2_j / T) - diag_i)
// One block per 16 rows; 8 waves sweep column tiles with WMMA, online softmax.
// ---------------------------------------------------------------------------
__global__ __launch_bounds__(256) void nce_kernel(
    const float* __restrict__ V1, const float* __restrict__ V2,
    int M, float* __restrict__ accum)
{
    __shared__ unsigned short As[16][64];
    __shared__ float red_m[8][16], red_s[8][16], red_d[8][16];
    const int tid = threadIdx.x, lane = tid & 31, wid = tid >> 5;
    const int lh = lane & 15, hh = lane >> 4;
    const int mbase = blockIdx.x * 16;

    for (int e = tid; e < 16 * 64; e += 256) {
        int m = e >> 6, k = e & 63;
        As[m][k] = f2bf(V1[(size_t)(mbase + m) * DM + k]);
    }
    __syncthreads();

    FragB fa0, fa1;
    {
        const unsigned int* ar = (const unsigned int*)&As[lh][0];
        const int kb2 = hh * 8;
        #pragma unroll
        for (int i = 0; i < 4; i++) {
            fa0.u[i] = ar[kb2 + i];      fa0.u[4 + i] = ar[4 + kb2 + i];
            fa1.u[i] = ar[16 + kb2 + i]; fa1.u[4 + i] = ar[20 + kb2 + i];
        }
    }

    float m_run[8], s_run[8], d_run[8];
    #pragma unroll
    for (int r = 0; r < 8; r++) { m_run[r] = -__builtin_inff(); s_run[r] = 0.f; d_run[r] = 0.f; }

    const int ntiles = M >> 4;
    for (int jt = wid; jt < ntiles; jt += 8) {
        const int jbase = jt * 16;
        // B = v2 tile transposed: scores = v1 @ v2^T, so B[k][n] = v2[j][k]
        FragB fb0, fb1;
        const float* vrow = &V2[(size_t)(jbase + lh) * DM + hh * 16];
        #pragma unroll
        for (int t = 0; t < 8; t++) {
            fb0.u[t] = pack2(vrow[2 * t],      vrow[2 * t + 1]);
            fb1.u[t] = pack2(vrow[32 + 2 * t], vrow[32 + 2 * t + 1]);
        }
        f32x8 c = {};
        c = __builtin_amdgcn_wmma_f32_16x16x32_bf16(false, fa0.v, false, fb0.v, (short)0, c, false, false);
        c = __builtin_amdgcn_wmma_f32_16x16x32_bf16(false, fa1.v, false, fb1.v, (short)0, c, false, false);
        #pragma unroll
        for (int r = 0; r < 8; r++) {
            float x = c[r] * INV_TEMP;
            int rowg = mbase + hh * 8 + r;
            int colg = jbase + lh;
            float tm = x;
            #pragma unroll
            for (int msk = 1; msk < 16; msk <<= 1) tm = fmaxf(tm, __shfl_xor(tm, msk, 32));
            float mnew = fmaxf(m_run[r], tm);
            float p = __expf(x - mnew);
            float ts = p;
            #pragma unroll
            for (int msk = 1; msk < 16; msk <<= 1) ts += __shfl_xor(ts, msk, 32);
            s_run[r] = s_run[r] * __expf(m_run[r] - mnew) + ts;
            m_run[r] = mnew;
            if (colg == rowg) d_run[r] += x;
        }
    }
    #pragma unroll
    for (int r = 0; r < 8; r++) {
        #pragma unroll
        for (int msk = 1; msk < 16; msk <<= 1) d_run[r] += __shfl_xor(d_run[r], msk, 32);
    }
    if (lh == 0) {
        const int base = hh * 8;
        #pragma unroll
        for (int r = 0; r < 8; r++) {
            red_m[wid][base + r] = m_run[r];
            red_s[wid][base + r] = s_run[r];
            red_d[wid][base + r] = d_run[r];
        }
    }
    __syncthreads();
    if (tid < 16) {
        float Mx = -__builtin_inff();
        #pragma unroll
        for (int w2 = 0; w2 < 8; w2++) Mx = fmaxf(Mx, red_m[w2][tid]);
        float S = 0.f, Dg = 0.f;
        #pragma unroll
        for (int w2 = 0; w2 < 8; w2++) {
            S  += red_s[w2][tid] * __expf(red_m[w2][tid] - Mx);
            Dg += red_d[w2][tid];
        }
        float part = (__logf(S) + Mx) - Dg;
        #pragma unroll
        for (int msk = 1; msk < 16; msk <<= 1) part += __shfl_xor(part, msk, 32);
        if (tid == 0) atomicAdd(accum, part);
    }
}

// ---------------------------------------------------------------------------
// Small helpers (float4 where counts are divisible by 4)
// ---------------------------------------------------------------------------
__global__ void vec_zero4(float4* p, int n4) {
    int i = blockIdx.x * 256 + threadIdx.x;
    if (i < n4) p[i] = make_float4(0.f, 0.f, 0.f, 0.f);
}
__global__ void vec_add4(float4* a, const float4* __restrict__ b, int n4) {
    int i = blockIdx.x * 256 + threadIdx.x;
    if (i < n4) {
        float4 x = a[i], y = b[i];
        a[i] = make_float4(x.x + y.x, x.y + y.y, x.z + y.z, x.w + y.w);
    }
}
__global__ void vec_scale4(float4* __restrict__ o, const float4* __restrict__ a, float s, int n4) {
    int i = blockIdx.x * 256 + threadIdx.x;
    if (i < n4) {
        float4 x = a[i];
        o[i] = make_float4(x.x * s, x.y * s, x.z * s, x.w * s);
    }
}
__global__ void vec_zero(float* p, int n) {
    int i = blockIdx.x * 256 + threadIdx.x; if (i < n) p[i] = 0.f;
}
// row dot for disc layer 3 (256 -> 1); one wave per row
__global__ __launch_bounds__(256) void rowdot(
    const float* __restrict__ A, const float* __restrict__ w,
    const float* __restrict__ b, float* __restrict__ outp, int M, int K)
{
    int row = blockIdx.x * 8 + (threadIdx.x >> 5);
    int lane = threadIdx.x & 31;
    if (row >= M) return;
    float s = 0.f;
    for (int k = lane; k < K; k += 32) s += A[(size_t)row * K + k] * w[k];
    #pragma unroll
    for (int off = 16; off; off >>= 1) s += __shfl_xor(s, off, 32);
    if (lane == 0) outp[row] = s + b[0];
}
__global__ void finalize_loss(const float* __restrict__ lacc, float* __restrict__ out_loss) {
    if (blockIdx.x == 0 && threadIdx.x == 0) {
        float l = SSLW * (lacc[0] / (float)STU + lacc[1] / (float)EXERN)
                + SSLW * (lacc[2] / (float)STU + lacc[3] / (float)EXERN);
        *out_loss = l;
    }
}
__global__ void gather_out(
    const float* __restrict__ o1, const float* __restrict__ o2,
    const float* __restrict__ disc1, const float* __restrict__ disc_tab,
    const int* __restrict__ sid, const int* __restrict__ eid,
    float* __restrict__ outp, int B_)
{
    int gid = blockIdx.x * 256 + threadIdx.x;
    int seg1 = B_ * DM, seg2 = 2 * seg1, seg3 = seg2 + B_;
    int total = seg3 + KNOWN * DM;
    if (gid >= total) return;
    float r;
    if (gid < seg1) {
        int b = gid >> 6, d = gid & 63; size_t o = (size_t)sid[b] * DM + d;
        r = LAMW * o1[o] + (1.f - LAMW) * o2[o];
    } else if (gid < seg2) {
        int g = gid - seg1; int b = g >> 6, d = g & 63;
        size_t o = (size_t)(STU + eid[b]) * DM + d;
        r = LAMW * o1[o] + (1.f - LAMW) * o2[o];
    } else if (gid < seg3) {
        int b = gid - seg2; int e = eid[b];
        r = LAMW * disc1[e] + (1.f - LAMW) * disc_tab[e];
    } else {
        int g = gid - seg3; int kk = g >> 6, d = g & 63;
        size_t o = (size_t)(STU + EXERN + kk) * DM + d;
        r = LAMW * o1[o] + (1.f - LAMW) * o2[o];
    }
    outp[gid] = r;
}

// ---------------------------------------------------------------------------
extern "C" void kernel_launch(void* const* d_in, const int* in_sizes, int n_in,
                              void* d_out, int out_size, void* d_ws, size_t ws_size,
                              hipStream_t stream)
{
    const float* text_stu  = (const float*)d_in[0];
    const float* text_exer = (const float*)d_in[1];
    const float* text_know = (const float*)d_in[2];
    const float* mlp_W1 = (const float*)d_in[3];
    const float* mlp_b1 = (const float*)d_in[4];
    const float* mlp_p1 = (const float*)d_in[5];
    const float* mlp_W2 = (const float*)d_in[6];
    const float* mlp_b2 = (const float*)d_in[7];
    const float* mlp_p2 = (const float*)d_in[8];
    const float* mlp_W3 = (const float*)d_in[9];
    const float* mlp_b3 = (const float*)d_in[10];
    const float* disc_W1 = (const float*)d_in[11];
    const float* disc_b1 = (const float*)d_in[12];
    const float* disc_p1 = (const float*)d_in[13];
    const float* disc_W2 = (const float*)d_in[14];
    const float* disc_b2 = (const float*)d_in[15];
    const float* disc_p2 = (const float*)d_in[16];
    const float* disc_W3 = (const float*)d_in[17];
    const float* disc_b3 = (const float*)d_in[18];
    const float* stu_tab  = (const float*)d_in[19];
    const float* exer_tab = (const float*)d_in[20];
    const float* know_tab = (const float*)d_in[21];
    const float* disc_tab = (const float*)d_in[22];
    const float* concat_W = (const float*)d_in[23];
    const float* concat_b = (const float*)d_in[24];
    const float* right_val  = (const float*)d_in[25];
    const int*   right_idx  = (const int*)d_in[26];
    const float* wrong_val  = (const float*)d_in[27];
    const int*   wrong_idx  = (const int*)d_in[28];
    const float* rightf_val = (const float*)d_in[29];
    const int*   rightf_idx = (const int*)d_in[30];
    const float* wrongf_val = (const float*)d_in[31];
    const int*   wrongf_idx = (const int*)d_in[32];
    const int* student_id  = (const int*)d_in[33];
    const int* exercise_id = (const int*)d_in[34];
    const int E_ = in_sizes[25];
    const int B_ = in_sizes[33];

    // workspace layout
    float* wsp = (float*)d_ws;
    size_t off = 0;
    auto alloc = [&](size_t n) { float* p = wsp + off; off += n; return p; };
    float* all1 = alloc(ND);
    float* all2 = alloc(ND);
    float* o1   = alloc(ND);
    float* o1f  = alloc(ND);
    float* o2   = alloc(ND);
    float* o2f  = alloc(ND);
    float* bufA = alloc(ND);
    float* bufB = alloc(ND);
    float* bufC = alloc(ND);
    float* bufD = alloc(ND);
    float* acc  = alloc(ND);
    float* cur  = alloc(ND);
    float* h512 = alloc((size_t)STU * 512);
    float* h256 = alloc((size_t)STU * 256);
    float* disc1 = alloc(EXERN);
    float* lacc  = alloc(4);

    auto gemm = [&](const float* A0, const float* A1, int Ksplit, int lda0, int lda1,
                    const float* W, const float* bias, const float* slope,
                    float* Cp, int M, int K, int Nout) {
        dim3 g(Nout / 64, (M + 63) / 64);
        gemm_bias_act<<<g, 256, 0, stream>>>(A0, A1, Ksplit, lda0, lda1, W, bias, slope, Cp, M, K, Nout);
    };
    auto zero4 = [&](float* p, size_t n) {  // n divisible by 4
        int n4 = (int)(n / 4);
        vec_zero4<<<(n4 + 255) / 256, 256, 0, stream>>>((float4*)p, n4);
    };
    auto spmm = [&](const int* idx, const float* val, const float* emb, float* outp) {
        long total = (long)E_ * 64;
        spmm_kernel<<<(int)((total + 255) / 256), 256, 0, stream>>>(idx, val, emb, outp, E_);
    };

    // ---- text MLPs -> all1 ----
    auto mlp3 = [&](const float* X, int M, int i, float* dst) {
        gemm(X, nullptr, 768, 768, 0, mlp_W1 + (size_t)i * 768 * 512, mlp_b1 + i * 512, mlp_p1 + i, h512, M, 768, 512);
        gemm(h512, nullptr, 512, 512, 0, mlp_W2 + (size_t)i * 512 * 256, mlp_b2 + i * 256, mlp_p2 + i, h256, M, 512, 256);
        gemm(h256, nullptr, 256, 256, 0, mlp_W3 + (size_t)i * 256 * 64, mlp_b3 + i * 64, nullptr, dst, M, 256, 64);
    };
    mlp3(text_stu,  STU,   0, all1);
    mlp3(text_exer, EXERN, 1, all1 + (size_t)STU * DM);
    mlp3(text_know, KNOWN, 2, all1 + (size_t)(STU + EXERN) * DM);

    // ---- all2 = concat of free tables ----
    hipMemcpyAsync(all2, stu_tab, (size_t)STU * DM * 4, hipMemcpyDeviceToDevice, stream);
    hipMemcpyAsync(all2 + (size_t)STU * DM, exer_tab, (size_t)EXERN * DM * 4, hipMemcpyDeviceToDevice, stream);
    hipMemcpyAsync(all2 + (size_t)(STU + EXERN) * DM, know_tab, (size_t)KNOWN * DM * 4, hipMemcpyDeviceToDevice, stream);

    // ---- disc MLP over exercises ----
    gemm(text_exer, nullptr, 768, 768, 0, disc_W1, disc_b1, disc_p1, h512, EXERN, 768, 512);
    gemm(h512, nullptr, 512, 512, 0, disc_W2, disc_b2, disc_p2, h256, EXERN, 512, 256);
    rowdot<<<(EXERN + 7) / 8, 256, 0, stream>>>(h256, disc_W3, disc_b3, disc1, EXERN, 256);

    // ---- GCN conv: 3 layers, mean over 4 embeddings ----
    auto conv = [&](const float* embIn, const int* ridx, const float* rval,
                    const int* widx, const float* wval, float* oOut) {
        float* rb[2] = { bufA, bufB };
        float* wb[2] = { bufC, bufD };
        hipMemcpyAsync(rb[0], embIn, ND * 4, hipMemcpyDeviceToDevice, stream);
        hipMemcpyAsync(wb[0], embIn, ND * 4, hipMemcpyDeviceToDevice, stream);
        hipMemcpyAsync(acc,   embIn, ND * 4, hipMemcpyDeviceToDevice, stream);
        int c = 0;
        for (int l = 0; l < 3; l++) {
            zero4(rb[1 - c], ND); spmm(ridx, rval, rb[c], rb[1 - c]);
            zero4(wb[1 - c], ND); spmm(widx, wval, wb[c], wb[1 - c]);
            c = 1 - c;
            // cur = concat(r, w) @ concat_W + concat_b
            gemm(rb[c], wb[c], 64, 64, 64, concat_W, concat_b, nullptr, cur, NTOT, 128, 64);
            int n4 = (int)(ND / 4);
            vec_add4<<<(n4 + 255) / 256, 256, 0, stream>>>((float4*)acc, (const float4*)cur, n4);
        }
        int n4 = (int)(ND / 4);
        vec_scale4<<<(n4 + 255) / 256, 256, 0, stream>>>((float4*)oOut, (const float4*)acc, 0.25f, n4);
    };
    conv(all1, right_idx,  right_val,  wrong_idx,  wrong_val,  o1);
    conv(all1, rightf_idx, rightf_val, wrongf_idx, wrongf_val, o1f);
    conv(all2, right_idx,  right_val,  wrong_idx,  wrong_val,  o2);
    conv(all2, rightf_idx, rightf_val, wrongf_idx, wrongf_val, o2f);

    // ---- InfoNCE terms ----
    vec_zero<<<1, 64, 0, stream>>>(lacc, 4);
    nce_kernel<<<STU / 16,   256, 0, stream>>>(o1,                     o1f,                     STU,   lacc + 0);
    nce_kernel<<<EXERN / 16, 256, 0, stream>>>(o1 + (size_t)STU * DM,  o1f + (size_t)STU * DM,  EXERN, lacc + 1);
    nce_kernel<<<STU / 16,   256, 0, stream>>>(o2,                     o2f,                     STU,   lacc + 2);
    nce_kernel<<<EXERN / 16, 256, 0, stream>>>(o2 + (size_t)STU * DM,  o2f + (size_t)STU * DM,  EXERN, lacc + 3);

    // ---- outputs ----
    float* out = (float*)d_out;
    int loss_off = 2 * B_ * DM + B_ + KNOWN * DM;
    finalize_loss<<<1, 64, 0, stream>>>(lacc, out + loss_off);
    int total = loss_off;
    gather_out<<<(total + 255) / 256, 256, 0, stream>>>(o1, o2, disc1, disc_tab,
                                                        student_id, exercise_id, out, B_);
}
